// MultiHead_65481071410966
// MI455X (gfx1250) — matrix operand order repaired and checked by reading, not compiled, not run on hardware
//
#include <hip/hip_runtime.h>
#include <hip/hip_bf16.h>

// ---------------------------------------------------------------------------
// MI455X (gfx1250) multi-head attention, bf16 WMMA pipeline, wave32.
// Async (ASYNCcnt) global->LDS staging for K/V in the attention kernel.
// ---------------------------------------------------------------------------

typedef __attribute__((ext_vector_type(16))) __bf16 v16bf;
typedef __attribute__((ext_vector_type(8)))  __bf16 v8bf;
typedef __attribute__((ext_vector_type(8)))  float  v8f;
typedef __attribute__((ext_vector_type(4)))  float  v4f;
typedef __attribute__((ext_vector_type(4)))  unsigned v4u;

#define D_MODEL  1024
#define N_HEAD   16
#define HEAD_DIM 64
#define SEQ      2048
#define BATCH    2
#define MROWS    (BATCH * SEQ)   // 4096 rows for the projection GEMMs
#define KSTR     72              // padded LDS row stride (bf16 elements)

// ---- helpers --------------------------------------------------------------

// cheap truncating f32 -> bf16
__device__ __forceinline__ __bf16 f2bf(float f) {
  unsigned u;
  __builtin_memcpy(&u, &f, 4);
  unsigned short h = (unsigned short)(u >> 16);
  __bf16 b;
  __builtin_memcpy(&b, &h, 2);
  return b;
}

__device__ __forceinline__ v8f wmma_bf16(v16bf a, v16bf b, v8f c) {
  // v_wmma_f32_16x16x32_bf16 : D = A(16x32) * B(32x16) + C
  return __builtin_amdgcn_wmma_f32_16x16x32_bf16(
      /*neg_a=*/false, a, /*neg_b=*/false, b,
      /*c_mod=*/(short)0, c, /*reuse_a=*/false, /*reuse_b=*/false);
}

// A-fragment (16x32) from bf16 memory (global or LDS).
// lane row = l&15;  elements 0..7 = K 8*hb+0..7, 8..15 = K 16+8*hb+0..7
__device__ __forceinline__ v16bf a_frag_bf16(const __bf16* rowptr, int hb) {
  v8bf b0 = *(const v8bf*)(rowptr + 8 * hb);
  v8bf b1 = *(const v8bf*)(rowptr + 16 + 8 * hb);
  return __builtin_shufflevector(b0, b1, 0, 1, 2, 3, 4, 5, 6, 7,
                                 8, 9, 10, 11, 12, 13, 14, 15);
}

// B-fragment (32x16).  lane col = l&15; elements 0..7 = K 8*hb+0..7, 8..15 = K 16+8*hb+0..7.
__device__ __forceinline__ v16bf b_frag_bf16(const __bf16* colptr, int hb) {
  v8bf b0 = *(const v8bf*)(colptr + 8 * hb);
  v8bf b1 = *(const v8bf*)(colptr + 16 + 8 * hb);
  return __builtin_shufflevector(b0, b1, 0, 1, 2, 3, 4, 5, 6, 7,
                                 8, 9, 10, 11, 12, 13, 14, 15);
}

__device__ __forceinline__ void lds_fence() {
  __asm__ volatile("s_wait_dscnt 0" ::: "memory");
}

// async DMA: 16 bytes global -> LDS, tracked by ASYNCcnt (no VGPR data)
__device__ __forceinline__ void async_b128(unsigned lds_addr, const __bf16* g) {
  __asm__ volatile("global_load_async_to_lds_b128 %0, %1, off"
                   :: "v"(lds_addr), "v"(g) : "memory");
}
__device__ __forceinline__ void wait_async4() {
  __asm__ volatile("s_wait_asynccnt 4" ::: "memory");
}
__device__ __forceinline__ void wait_async0() {
  __asm__ volatile("s_wait_asynccnt 0" ::: "memory");
}

// ---- kernel 0: bulk f32 -> bf16 (v_perm_b32 pair packing, b128 in/out) ----

__global__ __launch_bounds__(256)
void k_cvt_bf16(const float* __restrict__ in, __bf16* __restrict__ outp) {
  int i = (blockIdx.x * 256 + threadIdx.x) * 8;
  const v4u* p = (const v4u*)(in + i);
  v4u a = p[0], b = p[1];
  v4u o;
  o.x = __builtin_amdgcn_perm(a.y, a.x, 0x07060302u);  // {hi16(a1),hi16(a0)}
  o.y = __builtin_amdgcn_perm(a.w, a.z, 0x07060302u);
  o.z = __builtin_amdgcn_perm(b.y, b.x, 0x07060302u);
  o.w = __builtin_amdgcn_perm(b.w, b.z, 0x07060302u);
  *(v4u*)(outp + i) = o;
}

// ---- kernel 1: weight transpose + f32->bf16  (W[k][n] -> Wt[n][k]) --------

__global__ __launch_bounds__(256)
void k_transpose_w(const float* __restrict__ w, __bf16* __restrict__ wt) {
  __shared__ float tile[16][17];
  int tx = threadIdx.x & 15, ty = threadIdx.x >> 4;
  int bx = blockIdx.x, by = blockIdx.y;           // bx: n-tile, by: k-tile
  tile[ty][tx] = w[(by * 16 + ty) * D_MODEL + bx * 16 + tx];
  __syncthreads();
  wt[(bx * 16 + ty) * D_MODEL + by * 16 + tx] = f2bf(tile[tx][ty]);
}

// ---- kernel 2: projection GEMM  act(bf16)[4096,1024] @ Wt(bf16) + bias ----
// vmode 0: out[((b*H + h)*SEQ + s)*Hd + d]   (Q / K layout)
// vmode 1: out[((b*H + h)*Hd + d)*SEQ + s]   (V transposed layout)

__global__ __launch_bounds__(256)
void k_proj(const __bf16* __restrict__ act, const __bf16* __restrict__ wT,
            const float* __restrict__ bias, __bf16* __restrict__ out,
            int vmode) {
  int tid = threadIdx.x;
  int wave = tid >> 5, l = tid & 31, hb = l >> 4, ln = l & 15;
  int task = blockIdx.x * 8 + wave;  // 4096 wave tasks
  int m0 = (task >> 4) * 16;         // row tile base (0..4080)
  int n0 = (task & 15) * 64;         // col group base (0..960)

  v8f acc[4];
#pragma unroll
  for (int t = 0; t < 4; ++t) acc[t] = (v8f){0.f,0.f,0.f,0.f,0.f,0.f,0.f,0.f};

  const __bf16* arow = act + (size_t)(m0 + ln) * D_MODEL;
  for (int k0 = 0; k0 < D_MODEL; k0 += 32) {
    __builtin_prefetch(arow + k0 + 512, 0, 3);
    v16bf a = a_frag_bf16(arow + k0, hb);
#pragma unroll
    for (int t = 0; t < 4; ++t) {
      const __bf16* bp = wT + (size_t)(n0 + t * 16 + ln) * D_MODEL + k0;
      acc[t] = wmma_bf16(a, b_frag_bf16(bp, hb), acc[t]);
    }
  }

#pragma unroll
  for (int t = 0; t < 4; ++t) {
    int n = n0 + t * 16 + ln;
    float bv = bias[n];
    int h = n >> 6, d = n & 63;
#pragma unroll
    for (int v = 0; v < 8; ++v) {
      int m = m0 + v + 8 * hb;
      int b = m >> 11, s = m & 2047;
      float val = acc[t][v] + bv;
      int idx = vmode ? (((b * N_HEAD + h) * HEAD_DIM + d) * SEQ + s)
                      : (((b * N_HEAD + h) * SEQ + s) * HEAD_DIM + d);
      out[idx] = f2bf(val);
    }
  }
}

// ---- kernel 3: flash attention --------------------------------------------
// 8 waves/block share one (b,h); K/V 64-key blocks are staged to LDS with
// double-buffered global_load_async_to_lds_b128 and consumed by all waves.

__global__ __launch_bounds__(256)
void k_attn(const __bf16* __restrict__ Qp, const __bf16* __restrict__ Kp,
            const __bf16* __restrict__ Vt, const int* __restrict__ mask,
            __bf16* __restrict__ ctx) {
  __shared__ __align__(16) __bf16 kbuf[2][64 * KSTR];  // [key][d], 9 KB each
  __shared__ __align__(16) __bf16 vbuf[2][64 * KSTR];  // [d][key], 9 KB each
  __shared__ __align__(16) __bf16 plds[8 * 16 * KSTR]; // per-wave probs 16x64

  int tid = threadIdx.x;
  int wave = tid >> 5, l = tid & 31, hb = l >> 4, ln = l & 15;
  int bh = blockIdx.x >> 4;                 // all 8 waves share (b,h)
  int qt = (blockIdx.x * 8 + wave) & 127;   // per-wave query tile
  int b = bh >> 4, h = bh & 15;
  int q0 = qt * 16;
  const float inv = 0.12499999984375f;      // 1/(sqrt(64)+1e-8)

  // staging coords: this thread moves 2 x b128 of K and 2 x b128 of V
  int srow = tid >> 2;         // 0..63
  int scol = (tid & 3) * 16;   // 0,16,32,48

  // Q fragments (d = 0..31 and 32..63)
  const __bf16* qrow = Qp + (size_t)(bh * SEQ + q0 + ln) * HEAD_DIM;
  v16bf qa0 = a_frag_bf16(qrow, hb);
  v16bf qa1 = a_frag_bf16(qrow + 32, hb);

  float mrow[8], lrow[8];
  v8f o[4];
#pragma unroll
  for (int v = 0; v < 8; ++v) { mrow[v] = -1e30f; lrow[v] = 0.f; }
#pragma unroll
  for (int t = 0; t < 4; ++t) o[t] = (v8f){0.f,0.f,0.f,0.f,0.f,0.f,0.f,0.f};

  const int* mbase = mask + (size_t)(b * SEQ + q0) * SEQ;
  __bf16* myp = plds + wave * 16 * KSTR;

  auto stage = [&](int buf, int kb) {
    unsigned ka = (unsigned)(size_t)&kbuf[buf][srow * KSTR + scol];
    unsigned va = (unsigned)(size_t)&vbuf[buf][srow * KSTR + scol];
    const __bf16* kg = Kp + (size_t)(bh * SEQ + kb + srow) * HEAD_DIM + scol;
    const __bf16* vg = Vt + (size_t)(bh * HEAD_DIM + srow) * SEQ + kb + scol;
    async_b128(ka, kg);       async_b128(ka + 16, kg + 8);
    async_b128(va, vg);       async_b128(va + 16, vg + 8);
  };

  stage(0, 0);

  for (int kb = 0; kb < SEQ; kb += 64) {
    int cur = (kb >> 6) & 1;
    if (kb + 64 < SEQ) {
      stage(cur ^ 1, kb + 64);  // prefetch next block (async)
      wait_async4();            // current block's 4 ops are complete
    } else {
      wait_async0();
    }
    __syncthreads();            // all waves' fills visible

    const __bf16* ks = kbuf[cur];
    const __bf16* vs = vbuf[cur];

    float p[4][8];
    // scores for four 16-key tiles (K from LDS)
#pragma unroll
    for (int t = 0; t < 4; ++t) {
      const __bf16* krow = ks + (t * 16 + ln) * KSTR;
      v8f sc = (v8f){0.f,0.f,0.f,0.f,0.f,0.f,0.f,0.f};
      sc = wmma_bf16(qa0, b_frag_bf16(krow, hb), sc);
      sc = wmma_bf16(qa1, b_frag_bf16(krow + 32, hb), sc);
#pragma unroll
      for (int v = 0; v < 8; ++v) {
        int mv = mbase[(v + 8 * hb) * SEQ + kb + t * 16 + ln];
        p[t][v] = sc[v] * inv + (float)mv * (-1e9f);
      }
    }
    // online softmax per row (row v+8*hb spans the 16-lane column group)
#pragma unroll
    for (int v = 0; v < 8; ++v) {
      float rm = fmaxf(fmaxf(p[0][v], p[1][v]), fmaxf(p[2][v], p[3][v]));
      for (int off = 1; off < 16; off <<= 1)
        rm = fmaxf(rm, __shfl_xor(rm, off, 32));
      float mnew = fmaxf(mrow[v], rm);
      float alpha = __expf(mrow[v] - mnew);
      float rs = 0.f;
#pragma unroll
      for (int t = 0; t < 4; ++t) {
        p[t][v] = __expf(p[t][v] - mnew);
        rs += p[t][v];
      }
      for (int off = 1; off < 16; off <<= 1)
        rs += __shfl_xor(rs, off, 32);
      lrow[v] = lrow[v] * alpha + rs;
      mrow[v] = mnew;
#pragma unroll
      for (int t = 0; t < 4; ++t) o[t][v] *= alpha;
    }
    // C-layout probs -> LDS -> A-fragment layout (16 x 64)
#pragma unroll
    for (int t = 0; t < 4; ++t)
#pragma unroll
      for (int v = 0; v < 8; ++v)
        myp[(v + 8 * hb) * KSTR + t * 16 + ln] = f2bf(p[t][v]);
    lds_fence();
    v16bf pa0 = a_frag_bf16(myp + ln * KSTR, hb);       // keys kb+0..31
    v16bf pa1 = a_frag_bf16(myp + ln * KSTR + 32, hb);  // keys kb+32..63
    // P (16x64) @ V (64x64) with V from LDS ([d][key])
#pragma unroll
    for (int t = 0; t < 4; ++t) {
      const __bf16* vrow = vs + (t * 16 + ln) * KSTR;
      o[t] = wmma_bf16(pa0, b_frag_bf16(vrow, hb), o[t]);
      o[t] = wmma_bf16(pa1, b_frag_bf16(vrow + 32, hb), o[t]);
    }
    __syncthreads();            // everyone done reading before refill
  }

  // epilogue: normalize and write ctx[b][s][h*64+d] in bf16
#pragma unroll
  for (int t = 0; t < 4; ++t) {
    int col = h * HEAD_DIM + t * 16 + ln;
#pragma unroll
    for (int v = 0; v < 8; ++v) {
      int m = b * SEQ + q0 + v + 8 * hb;
      ctx[(size_t)m * D_MODEL + col] = f2bf(o[t][v] / lrow[v]);
    }
  }
}

// ---- kernel 4: output projection  ctx(bf16) @ WoT + bo -> f32 out ---------

__global__ __launch_bounds__(256)
void k_oproj(const __bf16* __restrict__ ctx, const __bf16* __restrict__ wT,
             const float* __restrict__ bias, float* __restrict__ out) {
  int tid = threadIdx.x;
  int wave = tid >> 5, l = tid & 31, hb = l >> 4, ln = l & 15;
  int task = blockIdx.x * 8 + wave;
  int m0 = (task >> 4) * 16;
  int n0 = (task & 15) * 64;

  v8f acc[4];
#pragma unroll
  for (int t = 0; t < 4; ++t) acc[t] = (v8f){0.f,0.f,0.f,0.f,0.f,0.f,0.f,0.f};

  const __bf16* arow = ctx + (size_t)(m0 + ln) * D_MODEL;
  for (int k0 = 0; k0 < D_MODEL; k0 += 32) {
    v16bf a = a_frag_bf16(arow + k0, hb);
#pragma unroll
    for (int t = 0; t < 4; ++t) {
      const __bf16* bp = wT + (size_t)(n0 + t * 16 + ln) * D_MODEL + k0;
      acc[t] = wmma_bf16(a, b_frag_bf16(bp, hb), acc[t]);
    }
  }

#pragma unroll
  for (int t = 0; t < 4; ++t) {
    int n = n0 + t * 16 + ln;
    float bv = bias[n];
#pragma unroll
    for (int v = 0; v < 8; ++v) {
      int m = m0 + v + 8 * hb;
      out[(size_t)m * D_MODEL + n] = acc[t][v] + bv;
    }
  }
}

// ---- launcher -------------------------------------------------------------
// Workspace layout (bytes), total 64 MB:
//   [0,8M)   : wqT,wkT,wvT,woT  bf16 [1024][1024] each (2 MB each)
//   [8M,16M) : Qp  bf16 [B][H][S][Hd]
//   [16M,24M): Kp  bf16 [B][H][S][Hd]
//   [24M,32M): Vt  bf16 [B][H][Hd][S]
//   [32M,40M): ctx bf16 [B][S][D]
//   [40M,64M): qb, kb, vb  bf16 [B][S][D]  (converted activations)

extern "C" void kernel_launch(void* const* d_in, const int* in_sizes, int n_in,
                              void* d_out, int out_size, void* d_ws,
                              size_t ws_size, hipStream_t stream) {
  const float* q    = (const float*)d_in[0];
  const float* k    = (const float*)d_in[1];
  const float* v    = (const float*)d_in[2];
  const int*   mask = (const int*)d_in[3];
  const float* wq = (const float*)d_in[4];
  const float* bq = (const float*)d_in[5];
  const float* wk = (const float*)d_in[6];
  const float* bk = (const float*)d_in[7];
  const float* wv = (const float*)d_in[8];
  const float* bv = (const float*)d_in[9];
  const float* wo = (const float*)d_in[10];
  const float* bo = (const float*)d_in[11];
  float* out = (float*)d_out;

  char* ws = (char*)d_ws;
  const size_t WT = (size_t)D_MODEL * D_MODEL * 2;                 // 2 MB
  const size_t PR = (size_t)BATCH * N_HEAD * SEQ * HEAD_DIM * 2;   // 8 MB
  __bf16* wqT = (__bf16*)(ws);
  __bf16* wkT = (__bf16*)(ws + WT);
  __bf16* wvT = (__bf16*)(ws + 2 * WT);
  __bf16* woT = (__bf16*)(ws + 3 * WT);
  __bf16* Qp  = (__bf16*)(ws + 4 * WT);
  __bf16* Kp  = (__bf16*)(ws + 4 * WT + PR);
  __bf16* Vt  = (__bf16*)(ws + 4 * WT + 2 * PR);
  __bf16* ctx = (__bf16*)(ws + 4 * WT + 3 * PR);
  __bf16* qb  = (__bf16*)(ws + 4 * WT + 4 * PR);
  __bf16* kb2 = (__bf16*)(ws + 4 * WT + 5 * PR);
  __bf16* vb2 = (__bf16*)(ws + 4 * WT + 6 * PR);

  // activations f32 -> bf16 (4M elements each, 8 per thread)
  int cvt_blocks = (MROWS * D_MODEL) / (256 * 8);  // 2048
  k_cvt_bf16<<<cvt_blocks, 256, 0, stream>>>(q, qb);
  k_cvt_bf16<<<cvt_blocks, 256, 0, stream>>>(k, kb2);
  k_cvt_bf16<<<cvt_blocks, 256, 0, stream>>>(v, vb2);

  dim3 tgrid(D_MODEL / 16, D_MODEL / 16);
  k_transpose_w<<<tgrid, 256, 0, stream>>>(wq, wqT);
  k_transpose_w<<<tgrid, 256, 0, stream>>>(wk, wkT);
  k_transpose_w<<<tgrid, 256, 0, stream>>>(wv, wvT);
  k_transpose_w<<<tgrid, 256, 0, stream>>>(wo, woT);

  // 4096 wave tasks / 8 waves per block = 512 blocks
  k_proj<<<512, 256, 0, stream>>>(qb, wqT, bq, Qp, 0);
  k_proj<<<512, 256, 0, stream>>>(kb2, wkT, bk, Kp, 0);
  k_proj<<<512, 256, 0, stream>>>(vb2, wvT, bv, Vt, 1);

  k_attn<<<512, 256, 0, stream>>>(Qp, Kp, Vt, mask, ctx);

  k_oproj<<<512, 256, 0, stream>>>(ctx, woT, bo, out);
}
